// GNN_63333587746868
// MI455X (gfx1250) — compile-verified, matrix operand order checked by reading, not run on hardware
//
#include <hip/hip_runtime.h>

#define DCH 256

typedef _Float16 half_t;
typedef half_t v16h __attribute__((ext_vector_type(16)));
typedef half_t v8h  __attribute__((ext_vector_type(8)));
typedef half_t v4h  __attribute__((ext_vector_type(4)));
typedef float  v8f  __attribute__((ext_vector_type(8)));
typedef float  v4f  __attribute__((ext_vector_type(4)));

// ---------------- degree / normalization ----------------
__global__ void k_deg_init(float* __restrict__ deg, int n) {
    int i = blockIdx.x * 256 + threadIdx.x;
    if (i < n) deg[i] = 1.0f;                      // self loop
}

__global__ void k_deg_accum(const int* __restrict__ dst, float* __restrict__ deg, int e) {
    int i = blockIdx.x * 256 + threadIdx.x;
    if (i < e) atomicAdd(&deg[dst[i]], 1.0f);
}

__global__ void k_deg_rsqrt(float* __restrict__ deg, int n) {
    int i = blockIdx.x * 256 + threadIdx.x;
    if (i < n) deg[i] = rsqrtf(deg[i]);            // deg >= 1 always
}

// ---------------- W (fp32 row-major KxN) -> f16 column-major (Wh[n*D+k]) ----
__global__ void k_conv_w(const float* __restrict__ W, half_t* __restrict__ Wh) {
    int g = blockIdx.x * 256 + threadIdx.x;        // g = n*256 + k
    int n = g >> 8, k = g & 255;
    Wh[g] = (half_t)W[k * DCH + n];
}

// ---------------- X (fp32) -> Ah (f16 row-major, zero-padded rows) ---------
__global__ void k_conv_x(const float* __restrict__ X, half_t* __restrict__ Ah, int n) {
    int g = blockIdx.x * 256 + threadIdx.x;        // g < Npad*256
    int row = g >> 8;
    Ah[g] = (row < n) ? (half_t)X[g] : (half_t)0.0f;
}

// ---------------- fragment load: two 16B chunks (K and K+16 halves) --------
__device__ __forceinline__ v16h ldfrag(const half_t* __restrict__ p) {
    v8h lo = *(const v8h*)p;
    v8h hi = *(const v8h*)(p + 16);
    v16h r;
#pragma unroll
    for (int i = 0; i < 8; ++i) { r[i] = lo[i]; r[8 + i] = hi[i]; }
    return r;
}

// ---------------- WMMA GEMM: H[Npad x 256] = Ah[Npad x 256] @ W ------------
// Software-pipelined: fragments for step k+1 are fetched before step k's WMMAs.
__global__ __launch_bounds__(256) void k_gemm_wmma(const half_t* __restrict__ Ah,
                                                   const half_t* __restrict__ Wh,
                                                   float* __restrict__ H) {
    const int lane  = threadIdx.x & 31;
    const int wave  = threadIdx.x >> 5;
    const int rbase = blockIdx.x * 32 + (wave >> 2) * 16;   // 16-row tile
    const int cbase = (wave & 3) * 64;                      // 64-col strip
    const int l15   = lane & 15;
    const int koff  = (lane < 16) ? 0 : 8;                  // K-half select

    const half_t* aptr  = Ah + (size_t)(rbase + l15) * DCH + koff;
    const half_t* bbase = Wh + (size_t)(cbase + l15) * DCH + koff;  // + j*16*DCH + k

    v8f acc[4] = {v8f{}, v8f{}, v8f{}, v8f{}};

    // prologue: fragments for k0 = 0
    v16h a = ldfrag(aptr);
    v16h b[4];
#pragma unroll
    for (int j = 0; j < 4; ++j) b[j] = ldfrag(bbase + (size_t)j * 16 * DCH);

#pragma unroll
    for (int k0 = 0; k0 < DCH; k0 += 32) {
        v16h an = a;
        v16h bn[4] = {b[0], b[1], b[2], b[3]};
        if (k0 + 32 < DCH) {                                 // compile-time after unroll
            an = ldfrag(aptr + k0 + 32);
#pragma unroll
            for (int j = 0; j < 4; ++j)
                bn[j] = ldfrag(bbase + (size_t)j * 16 * DCH + k0 + 32);
        }
#pragma unroll
        for (int j = 0; j < 4; ++j)
            acc[j] = __builtin_amdgcn_wmma_f32_16x16x32_f16(
                false, a, false, b[j], (short)0, acc[j], false, false);
        a = an;
#pragma unroll
        for (int j = 0; j < 4; ++j) b[j] = bn[j];
    }

    // C/D layout: VGPR i -> M = i (lanes 0-15) / 8+i (lanes 16-31), N = lane&15
    const int rstore = rbase + ((lane < 16) ? 0 : 8);
#pragma unroll
    for (int j = 0; j < 4; ++j) {
        int col = cbase + j * 16 + l15;
#pragma unroll
        for (int i = 0; i < 8; ++i)
            H[(size_t)(rstore + i) * DCH + col] = acc[j][i];   // padded rows ok
    }
}

// ---------------- self-loop seed: ACC = dinv^2 * H -------------------------
__global__ void k_self_init(const float* __restrict__ dinv, const float* __restrict__ H,
                            float* __restrict__ ACC, int n) {
    int g = blockIdx.x * 256 + threadIdx.x;
    int row = g >> 6;
    if (row >= n) return;
    int c = (g & 63) * 4;
    float nn = dinv[row]; nn *= nn;
    v4f h = *(const v4f*)(H + (size_t)row * DCH + c);
    v4f o = {nn * h[0], nn * h[1], nn * h[2], nn * h[3]};
    *(v4f*)(ACC + (size_t)row * DCH + c) = o;
}

// ---------------- edge scatter: one wave per edge --------------------------
__global__ __launch_bounds__(256) void k_edge(const int* __restrict__ ei,
                                              const float* __restrict__ dinv,
                                              const float* __restrict__ H,
                                              float* __restrict__ ACC, int e) {
    int g    = blockIdx.x * 256 + threadIdx.x;
    int edge = g >> 5;
    int lane = g & 31;
    if (edge >= e) return;
    int s = ei[edge];
    int d = ei[e + edge];
    float nrm = dinv[s] * dinv[d];
    const float* hp = H + (size_t)s * DCH;
    float*       ap = ACC + (size_t)d * DCH;
    int c = lane * 4;
    v4f h0 = *(const v4f*)(hp + c);
    v4f h1 = *(const v4f*)(hp + c + 128);
    atomicAdd(ap + c + 0,   nrm * h0[0]);
    atomicAdd(ap + c + 1,   nrm * h0[1]);
    atomicAdd(ap + c + 2,   nrm * h0[2]);
    atomicAdd(ap + c + 3,   nrm * h0[3]);
    atomicAdd(ap + c + 128, nrm * h1[0]);
    atomicAdd(ap + c + 129, nrm * h1[1]);
    atomicAdd(ap + c + 130, nrm * h1[2]);
    atomicAdd(ap + c + 131, nrm * h1[3]);
}

// ---------------- finalize layers 1-2: Ah = f16(relu(ACC + b)) -------------
__global__ void k_finalize_h(const float* __restrict__ ACC, const float* __restrict__ bias,
                             half_t* __restrict__ Ah, int n) {
    int g = blockIdx.x * 256 + threadIdx.x;
    int row = g >> 6;
    if (row >= n) return;
    int c = (g & 63) * 4;
    v4f v  = *(const v4f*)(ACC + (size_t)row * DCH + c);
    v4f bb = *(const v4f*)(bias + c);
    v4h o;
#pragma unroll
    for (int i = 0; i < 4; ++i) {
        float t = v[i] + bb[i];
        o[i] = (half_t)(t > 0.0f ? t : 0.0f);
    }
    *(v4h*)(Ah + (size_t)row * DCH + c) = o;
}

// ---------------- finalize layer 3: out = fp32(ACC + b) --------------------
__global__ void k_finalize_f(const float* __restrict__ ACC, const float* __restrict__ bias,
                             float* __restrict__ out, int n) {
    int g = blockIdx.x * 256 + threadIdx.x;
    int row = g >> 6;
    if (row >= n) return;
    int c = (g & 63) * 4;
    v4f v  = *(const v4f*)(ACC + (size_t)row * DCH + c);
    v4f bb = *(const v4f*)(bias + c);
    v4f o = {v[0] + bb[0], v[1] + bb[1], v[2] + bb[2], v[3] + bb[3]};
    *(v4f*)(out + (size_t)row * DCH + c) = o;
}

extern "C" void kernel_launch(void* const* d_in, const int* in_sizes, int n_in,
                              void* d_out, int out_size, void* d_ws, size_t ws_size,
                              hipStream_t stream) {
    const float* x  = (const float*)d_in[0];
    const int*   ei = (const int*)d_in[1];
    const float* Wm[3] = {(const float*)d_in[2], (const float*)d_in[4], (const float*)d_in[6]};
    const float* bm[3] = {(const float*)d_in[3], (const float*)d_in[5], (const float*)d_in[7]};
    float* out = (float*)d_out;

    const int N    = in_sizes[0] / DCH;
    const int E    = in_sizes[1] / 2;
    const int Npad = ((N + 31) / 32) * 32;

    // workspace carve-up (256B-aligned slices)
    char*  base = (char*)d_ws;
    size_t off  = 0;
    float*  deg = (float*)(base + off);  off += ((size_t)N * 4 + 255) & ~(size_t)255;
    half_t* Wh  = (half_t*)(base + off); off += (size_t)DCH * DCH * 2;
    half_t* Ah  = (half_t*)(base + off); off += (size_t)Npad * DCH * 2;
    float*  H   = (float*)(base + off);  off += (size_t)Npad * DCH * 4;
    float*  ACC = (float*)(base + off);  off += (size_t)Npad * DCH * 4;
    (void)ws_size; (void)n_in; (void)out_size;

    // normalization
    k_deg_init <<<(N + 255) / 256, 256, 0, stream>>>(deg, N);
    k_deg_accum<<<(E + 255) / 256, 256, 0, stream>>>(ei + E, deg, E);
    k_deg_rsqrt<<<(N + 255) / 256, 256, 0, stream>>>(deg, N);

    // layer-1 activation in f16 (with zero-padded tail rows)
    k_conv_x<<<Npad, 256, 0, stream>>>(x, Ah, N);

    for (int layer = 0; layer < 3; ++layer) {
        k_conv_w<<<(DCH * DCH) / 256, 256, 0, stream>>>(Wm[layer], Wh);
        k_gemm_wmma<<<Npad / 32, 256, 0, stream>>>(Ah, Wh, H);
        {
            long t = (long)N * 64;
            k_self_init<<<(unsigned)((t + 255) / 256), 256, 0, stream>>>(deg, H, ACC, N);
        }
        {
            long t = (long)E * 32;
            k_edge<<<(unsigned)((t + 255) / 256), 256, 0, stream>>>(ei, deg, H, ACC, E);
        }
        {
            long t = (long)N * 64;
            unsigned gsz = (unsigned)((t + 255) / 256);
            if (layer < 2)
                k_finalize_h<<<gsz, 256, 0, stream>>>(ACC, bm[layer], Ah, N);
            else
                k_finalize_f<<<gsz, 256, 0, stream>>>(ACC, bm[layer], out, N);
        }
    }
}
